// SelfReferentialTitans_83193516523560
// MI455X (gfx1250) — compile-verified, matrix operand order checked by reading, not run on hardware
//
#include <hip/hip_runtime.h>
#include <hip/hip_bf16.h>

// ---------------------------------------------------------------------------
// SelfReferentialTitans (D=C=1024) for MI455X / gfx1250 (wave32, WMMA).
// Phase A: time-parallel GEMMs via v_wmma_f32_16x16x32_bf16 with bf16x3
//          error-compensated split (hi*hi + hi*lo + lo*hi). Tiles are staged
//          to LDS *as bf16 hi/lo planes* (convert once), fragments come from
//          16B ds_load_b128s, and global loads of the next k-tile are
//          register-prefetched to overlap the WMMA burst.
// Phase B: 1024 sequential steps, 2 kernels/step over L2-resident 12MB state.
// Phase C: one final WMMA GEMM for the output projection.
// ---------------------------------------------------------------------------

#define DD 1024
#define CC 1024

typedef __bf16 bf16_t;
typedef bf16_t v16bf __attribute__((ext_vector_type(16)));
typedef bf16_t v8bf  __attribute__((ext_vector_type(8)));
typedef bf16_t v4bf  __attribute__((ext_vector_type(4)));
typedef float  v8f   __attribute__((ext_vector_type(8)));

__device__ __forceinline__ float silu_f(float x) {
    return x / (1.0f + __expf(-x));
}

// LDS plane row stride in bf16 elements: 32 data + 8 pad = 40 (80B).
// Keeps every 16B fragment block aligned and staggers banks (20*r mod 64).
#define LSTR 40

// Build one 16x32 bf16 fragment from an LDS plane.
// Frag element order per lane (ISA 7.12.2): elems 0..7 <- K 8*lh..8*lh+7,
// elems 8..15 <- K 16+8*lh..23+8*lh  => two contiguous 8-element runs.
__device__ __forceinline__ v16bf frag_from_lds(const bf16_t* plane, int row,
                                               int lh) {
    const bf16_t* p = plane + row * LSTR + 8 * lh;
    v8bf a0 = *(const v8bf*)(p);        // ds_load_b128
    v8bf a1 = *(const v8bf*)(p + 16);   // ds_load_b128
    return __builtin_shufflevector(a0, a1, 0, 1, 2, 3, 4, 5, 6, 7, 8, 9, 10,
                                   11, 12, 13, 14, 15);
}

// Convert float4 -> hi/lo bf16x4 and store to LDS planes (8B stores).
__device__ __forceinline__ void cvt_store(bf16_t* hi, bf16_t* lo, int idx,
                                          float4 a) {
    v4bf h, l;
    {
        bf16_t t;
        t = (bf16_t)a.x; h[0] = t; l[0] = (bf16_t)(a.x - (float)t);
        t = (bf16_t)a.y; h[1] = t; l[1] = (bf16_t)(a.y - (float)t);
        t = (bf16_t)a.z; h[2] = t; l[2] = (bf16_t)(a.z - (float)t);
        t = (bf16_t)a.w; h[3] = t; l[3] = (bf16_t)(a.w - (float)t);
    }
    *(v4bf*)(hi + idx) = h;
    *(v4bf*)(lo + idx) = l;
}

// ---------------------------------------------------------------------------
// Generic GEMM: Out[m,n] = act( sum_k A[m,k]*W[n,k] + bias[n] )
// M=N=K=1024. Block: 128 threads (4 waves). Tile 128(M) x 64(N), K-step 32.
// Wave w owns rows [32w, 32w+32): 2 m-subtiles x 4 n-tiles = 8 accumulators,
// 24 WMMAs per k-step per wave.
// ---------------------------------------------------------------------------
__global__ __launch_bounds__(128)
void gemm_bf16x3(const float* __restrict__ A, const float* __restrict__ W,
                 const float* __restrict__ bias, float* __restrict__ Out,
                 int act) {
    __shared__ __align__(16) bf16_t Ahi[128 * LSTR];
    __shared__ __align__(16) bf16_t Alo[128 * LSTR];
    __shared__ __align__(16) bf16_t Whi[64 * LSTR];
    __shared__ __align__(16) bf16_t Wlo[64 * LSTR];

    const int tid  = threadIdx.x;
    const int wave = tid >> 5;
    const int lane = tid & 31;
    const int lh   = lane >> 4;   // lane half
    const int l16  = lane & 15;
    const int m0 = blockIdx.y * 128;
    const int n0 = blockIdx.x * 64;

    v8f acc[2][4];
    #pragma unroll
    for (int s = 0; s < 2; ++s)
        #pragma unroll
        for (int j = 0; j < 4; ++j)
            #pragma unroll
            for (int e = 0; e < 8; ++e) acc[s][j][e] = 0.0f;

    // register prefetch buffers: A tile 128x32 = 1024 float4 -> 8/thread,
    // W tile 64x32 = 512 float4 -> 4/thread
    float4 ra[8], rw[4];

    // issue first tile's global loads
    #pragma unroll
    for (int i = 0; i < 8; ++i) {
        int s = tid + 128 * i, row = s >> 3, c4 = (s & 7) << 2;
        ra[i] = *(const float4*)(A + (size_t)(m0 + row) * DD + c4);
    }
    #pragma unroll
    for (int i = 0; i < 4; ++i) {
        int s = tid + 128 * i, row = s >> 3, c4 = (s & 7) << 2;
        rw[i] = *(const float4*)(W + (size_t)(n0 + row) * DD + c4);
    }

    for (int kt = 0; kt < DD / 32; ++kt) {
        __syncthreads();   // previous compute done reading LDS
        // stage current tile (convert once -> bf16 hi/lo planes)
        #pragma unroll
        for (int i = 0; i < 8; ++i) {
            int s = tid + 128 * i, row = s >> 3, c4 = (s & 7) << 2;
            cvt_store(Ahi, Alo, row * LSTR + c4, ra[i]);
        }
        #pragma unroll
        for (int i = 0; i < 4; ++i) {
            int s = tid + 128 * i, row = s >> 3, c4 = (s & 7) << 2;
            cvt_store(Whi, Wlo, row * LSTR + c4, rw[i]);
        }
        __syncthreads();

        // overlap: issue next tile's global loads before the WMMA burst
        if (kt + 1 < DD / 32) {
            int k0 = (kt + 1) * 32;
            #pragma unroll
            for (int i = 0; i < 8; ++i) {
                int s = tid + 128 * i, row = s >> 3, c4 = (s & 7) << 2;
                ra[i] = *(const float4*)(A + (size_t)(m0 + row) * DD + k0 + c4);
            }
            #pragma unroll
            for (int i = 0; i < 4; ++i) {
                int s = tid + 128 * i, row = s >> 3, c4 = (s & 7) << 2;
                rw[i] = *(const float4*)(W + (size_t)(n0 + row) * DD + k0 + c4);
            }
        }

        // fragments + 24 WMMAs
        v16bf ah[2], al[2];
        #pragma unroll
        for (int s = 0; s < 2; ++s) {
            int row = 32 * wave + 16 * s + l16;
            ah[s] = frag_from_lds(Ahi, row, lh);
            al[s] = frag_from_lds(Alo, row, lh);
        }
        #pragma unroll
        for (int j = 0; j < 4; ++j) {
            int nr = 16 * j + l16;
            v16bf bh = frag_from_lds(Whi, nr, lh);
            v16bf bl = frag_from_lds(Wlo, nr, lh);
            #pragma unroll
            for (int s = 0; s < 2; ++s) {
                acc[s][j] = __builtin_amdgcn_wmma_f32_16x16x32_bf16(
                    false, al[s], false, bh, (short)0, acc[s][j], false, false);
                acc[s][j] = __builtin_amdgcn_wmma_f32_16x16x32_bf16(
                    false, ah[s], false, bl, (short)0, acc[s][j], false, false);
                acc[s][j] = __builtin_amdgcn_wmma_f32_16x16x32_bf16(
                    false, ah[s], false, bh, (short)0, acc[s][j], false, false);
            }
        }
    }

    // epilogue: C/D layout -> n = l16, m = vi + 8*lh within a 16x16 tile
    #pragma unroll
    for (int s = 0; s < 2; ++s) {
        #pragma unroll
        for (int j = 0; j < 4; ++j) {
            int n = n0 + 16 * j + l16;
            float b = bias ? bias[n] : 0.0f;
            #pragma unroll
            for (int vi = 0; vi < 8; ++vi) {
                int m = m0 + 32 * wave + 16 * s + vi + 8 * lh;
                float v = acc[s][j][vi] + b;
                if (act) v = silu_f(v);
                Out[(size_t)m * DD + n] = v;
            }
        }
    }
}

// ---------------------------------------------------------------------------
// out[t] = scale * sigmoid( dot(H[t,:], w2) + b2 )  (eta: scale=0.01, alpha: 1)
// ---------------------------------------------------------------------------
__global__ __launch_bounds__(256)
void rowdot_sigmoid(const float* __restrict__ H, const float* __restrict__ w2,
                    const float* __restrict__ b2, float* __restrict__ outv,
                    float scale) {
    __shared__ float red[256];
    const int t = blockIdx.x;
    const float* h = H + (size_t)t * DD;
    float s = 0.0f;
    for (int i = threadIdx.x; i < DD; i += 256) s += h[i] * w2[i];
    red[threadIdx.x] = s;
    __syncthreads();
    for (int o = 128; o > 0; o >>= 1) {
        if (threadIdx.x < o) red[threadIdx.x] += red[threadIdx.x + o];
        __syncthreads();
    }
    if (threadIdx.x == 0) {
        float z = red[0] + b2[0];
        outv[t] = scale / (1.0f + __expf(-z));
    }
}

// ---------------------------------------------------------------------------
// Row-normalize K (y==0) and Q (y==1):  v /= max(||v||, 1e-12)
// ---------------------------------------------------------------------------
__global__ __launch_bounds__(256)
void rownorm2(float* __restrict__ Kv, float* __restrict__ Qv) {
    __shared__ float red[256];
    __shared__ float sc;
    const int t = blockIdx.x;
    float* v = (blockIdx.y == 0 ? Kv : Qv) + (size_t)t * DD;
    float s = 0.0f;
    for (int i = threadIdx.x; i < DD; i += 256) { float x = v[i]; s += x * x; }
    red[threadIdx.x] = s;
    __syncthreads();
    for (int o = 128; o > 0; o >>= 1) {
        if (threadIdx.x < o) red[threadIdx.x] += red[threadIdx.x + o];
        __syncthreads();
    }
    if (threadIdx.x == 0) sc = 1.0f / fmaxf(sqrtf(red[0]), 1e-12f);
    __syncthreads();
    float k = sc;
    for (int i = threadIdx.x; i < DD; i += 256) v[i] *= k;
}

// ---------------------------------------------------------------------------
// init: M_k = M_v = M_mem = W_base
// ---------------------------------------------------------------------------
__global__ __launch_bounds__(256)
void init_state(const float* __restrict__ Wb, float* __restrict__ MK,
                float* __restrict__ MV, float* __restrict__ MM) {
    size_t i = ((size_t)blockIdx.x * 256 + threadIdx.x) * 4;
    float4 w = *(const float4*)(Wb + i);
    *(float4*)(MK + i) = w;
    *(float4*)(MV + i) = w;
    *(float4*)(MM + i) = w;
}

// ---------------------------------------------------------------------------
// Step kernel 1: matvecs against current state (pre-update values).
// grid(128, 3): y = matrix {MK, MV, MM}; 8 waves/block, one row per wave.
// MM waves also compute o = MM@q, vhm = MM@v, and z_t = (o+v)*g.
// ---------------------------------------------------------------------------
__global__ __launch_bounds__(256)
void step_matvec(const float* __restrict__ MK, const float* __restrict__ MV,
                 const float* __restrict__ MM,
                 const float* __restrict__ Kn, const float* __restrict__ Qn,
                 const float* __restrict__ Vb, const float* __restrict__ Gg,
                 const float* __restrict__ VK, const float* __restrict__ VV,
                 float* __restrict__ yk, float* __restrict__ yv,
                 float* __restrict__ ym,
                 float* __restrict__ ek, float* __restrict__ ev,
                 float* __restrict__ em,
                 float* __restrict__ ovec, float* __restrict__ Z, int t) {
    const int wave = threadIdx.x >> 5;
    const int lane = threadIdx.x & 31;
    const int r    = blockIdx.x * 8 + wave;
    const int mat  = blockIdx.y;

    const float* Mrow = (mat == 0 ? MK : (mat == 1 ? MV : MM)) + (size_t)r * DD;
    const float* kr = Kn + (size_t)t * DD;

    float sk = 0.0f, sq = 0.0f, sv = 0.0f;
    if (mat == 2) {
        const float* qr = Qn + (size_t)t * DD;
        const float* vr = Vb + (size_t)t * DD;
        #pragma unroll
        for (int i = 0; i < 8; ++i) {
            int c = lane * 4 + i * 128;
            float4 m  = *(const float4*)(Mrow + c);
            float4 k4 = *(const float4*)(kr + c);
            float4 q4 = *(const float4*)(qr + c);
            float4 v4 = *(const float4*)(vr + c);
            sk += m.x * k4.x + m.y * k4.y + m.z * k4.z + m.w * k4.w;
            sq += m.x * q4.x + m.y * q4.y + m.z * q4.z + m.w * q4.w;
            sv += m.x * v4.x + m.y * v4.y + m.z * v4.z + m.w * v4.w;
        }
    } else {
        #pragma unroll
        for (int i = 0; i < 8; ++i) {
            int c = lane * 4 + i * 128;
            float4 m  = *(const float4*)(Mrow + c);
            float4 k4 = *(const float4*)(kr + c);
            sk += m.x * k4.x + m.y * k4.y + m.z * k4.z + m.w * k4.w;
        }
    }
    for (int off = 16; off > 0; off >>= 1) {
        sk += __shfl_xor(sk, off);
        sq += __shfl_xor(sq, off);
        sv += __shfl_xor(sv, off);
    }
    if (lane == 0) {
        if (mat == 0) {
            yk[r] = sk;
            ek[r] = sk - VK[(size_t)t * DD + r];
        } else if (mat == 1) {
            yv[r] = sk;
            ev[r] = sk - VV[(size_t)t * DD + r];
        } else {
            ym[r]   = sk;
            em[r]   = sk - sv;        // e_mem = MM@k - MM@v
            ovec[r] = sq;             // retrieval with pre-update MM
            float vt = Vb[(size_t)t * DD + r];
            float gt = Gg[(size_t)t * DD + r];
            Z[(size_t)t * DD + r] = (sq + vt) * gt;
        }
    }
}

// ---------------------------------------------------------------------------
// Step kernel 2: rank-1 decay update.
// M[r,:] = alpha*M[r,:] - eta*(y[r] + clip(e)[r]) * k[:]
// grid(1024, 3): one block per (row, matrix). Deterministic block-local
// reduction of ||e||^2 (no fp atomics).
// ---------------------------------------------------------------------------
__global__ __launch_bounds__(256)
void step_update(float* __restrict__ MK, float* __restrict__ MV,
                 float* __restrict__ MM, const float* __restrict__ Kn,
                 const float* __restrict__ yk, const float* __restrict__ yv,
                 const float* __restrict__ ym,
                 const float* __restrict__ ek, const float* __restrict__ ev,
                 const float* __restrict__ em,
                 const float* __restrict__ eta, const float* __restrict__ alpha,
                 int t) {
    __shared__ float red[256];
    __shared__ float ubc, abc;
    const int r   = blockIdx.x;
    const int mat = blockIdx.y;
    const int tid = threadIdx.x;

    float* M       = (mat == 0 ? MK : (mat == 1 ? MV : MM));
    const float* e = (mat == 0 ? ek : (mat == 1 ? ev : em));
    const float* y = (mat == 0 ? yk : (mat == 1 ? yv : ym));

    float s = 0.0f;
    for (int i = tid; i < DD; i += 256) { float x = e[i]; s += x * x; }
    red[tid] = s;
    __syncthreads();
    for (int o = 128; o > 0; o >>= 1) {
        if (tid < o) red[tid] += red[tid + o];
        __syncthreads();
    }
    if (tid == 0) {
        float n  = sqrtf(red[0]);
        float cs = (n > 1.0f) ? (1.0f / fmaxf(n, 1e-12f)) : 1.0f;  // grad clip
        ubc = eta[t] * (y[r] + cs * e[r]);
        abc = alpha[t];
    }
    __syncthreads();
    const float u = ubc, a = abc;
    const float* kr = Kn + (size_t)t * DD;
    float* Mrow = M + (size_t)r * DD;
    int c = tid * 4;
    float4 m  = *(const float4*)(Mrow + c);
    float4 k4 = *(const float4*)(kr + c);
    m.x = a * m.x - u * k4.x;
    m.y = a * m.y - u * k4.y;
    m.z = a * m.z - u * k4.z;
    m.w = a * m.w - u * k4.w;
    *(float4*)(Mrow + c) = m;
}

// ---------------------------------------------------------------------------
extern "C" void kernel_launch(void* const* d_in, const int* in_sizes, int n_in,
                              void* d_out, int out_size, void* d_ws,
                              size_t ws_size, hipStream_t stream) {
    (void)in_sizes; (void)n_in; (void)out_size; (void)ws_size;

    const float* X     = (const float*)d_in[0];
    const float* W_q   = (const float*)d_in[1];
    const float* Wk1   = (const float*)d_in[2];
    const float* bk1   = (const float*)d_in[3];
    const float* Wk2   = (const float*)d_in[4];
    const float* bk2   = (const float*)d_in[5];
    const float* Wv1   = (const float*)d_in[6];
    const float* bv1   = (const float*)d_in[7];
    const float* Wv2   = (const float*)d_in[8];
    const float* bv2   = (const float*)d_in[9];
    const float* We1   = (const float*)d_in[10];
    const float* be1   = (const float*)d_in[11];
    const float* We2   = (const float*)d_in[12];
    const float* be2   = (const float*)d_in[13];
    const float* Wa1   = (const float*)d_in[14];
    const float* ba1   = (const float*)d_in[15];
    const float* Wa2   = (const float*)d_in[16];
    const float* ba2   = (const float*)d_in[17];
    const float* Wbase = (const float*)d_in[18];
    const float* Wg    = (const float*)d_in[19];
    const float* bg    = (const float*)d_in[20];
    const float* Wp    = (const float*)d_in[21];
    const float* bp    = (const float*)d_in[22];

    float* ws = (float*)d_ws;
    const size_t MAT = (size_t)CC * DD;   // 1M floats
    float* H  = ws + 0 * MAT;   // MLP hidden scratch (phase A only)
    float* Kn = ws + 1 * MAT;   // normalized keys
    float* Vb = ws + 2 * MAT;   // values
    float* Qn = ws + 3 * MAT;   // normalized queries
    float* Gg = ws + 4 * MAT;   // gate = silu(Wg x + bg)
    float* VK = ws + 5 * MAT;   // v_hat_k  = MLP_k(v)
    float* VV = ws + 6 * MAT;   // v_hat_v  = MLP_v(v)
    float* MK = ws + 7 * MAT;   // fast weight M_k
    float* MV = ws + 8 * MAT;   // fast weight M_v
    float* MM = ws + 9 * MAT;   // fast weight M_mem
    float* Z  = H;              // alias: z_t collected in phase B
    float* sm   = ws + 10 * MAT;
    float* eta  = sm + 0 * CC;
    float* alph = sm + 1 * CC;
    float* yk   = sm + 2 * CC;
    float* yv   = sm + 3 * CC;
    float* ym   = sm + 4 * CC;
    float* ov   = sm + 5 * CC;
    float* ek   = sm + 6 * CC;
    float* ev   = sm + 7 * CC;
    float* em   = sm + 8 * CC;

    const dim3 gg(16, 8);  // 64(N) x 128(M) tiles over 1024x1024

    // ---- Phase A: batched WMMA GEMMs (time-parallel precompute) ----
    init_state<<<DD * DD / 1024, 256, 0, stream>>>(Wbase, MK, MV, MM);
    gemm_bf16x3<<<gg, 128, 0, stream>>>(X, Wk1, bk1, H, 1);
    gemm_bf16x3<<<gg, 128, 0, stream>>>(H, Wk2, bk2, Kn, 0);
    gemm_bf16x3<<<gg, 128, 0, stream>>>(X, Wv1, bv1, H, 1);
    gemm_bf16x3<<<gg, 128, 0, stream>>>(H, Wv2, bv2, Vb, 0);
    gemm_bf16x3<<<gg, 128, 0, stream>>>(X, We1, be1, H, 1);
    rowdot_sigmoid<<<CC, 256, 0, stream>>>(H, We2, be2, eta, 0.01f);
    gemm_bf16x3<<<gg, 128, 0, stream>>>(X, Wa1, ba1, H, 1);
    rowdot_sigmoid<<<CC, 256, 0, stream>>>(H, Wa2, ba2, alph, 1.0f);
    gemm_bf16x3<<<gg, 128, 0, stream>>>(X, W_q, nullptr, Qn, 0);
    gemm_bf16x3<<<gg, 128, 0, stream>>>(X, Wg, bg, Gg, 1);
    gemm_bf16x3<<<gg, 128, 0, stream>>>(Vb, Wk1, bk1, H, 1);
    gemm_bf16x3<<<gg, 128, 0, stream>>>(H, Wk2, bk2, VK, 0);
    gemm_bf16x3<<<gg, 128, 0, stream>>>(Vb, Wv1, bv1, H, 1);
    gemm_bf16x3<<<gg, 128, 0, stream>>>(H, Wv2, bv2, VV, 0);
    rownorm2<<<dim3(CC, 2), 256, 0, stream>>>(Kn, Qn);

    // ---- Phase B: sequential DGD scan over L2-resident state ----
    for (int t = 0; t < CC; ++t) {
        step_matvec<<<dim3(128, 3), 256, 0, stream>>>(
            MK, MV, MM, Kn, Qn, Vb, Gg, VK, VV,
            yk, yv, ym, ek, ev, em, ov, Z, t);
        step_update<<<dim3(DD, 3), 256, 0, stream>>>(
            MK, MV, MM, Kn, yk, yv, ym, ek, ev, em, eta, alph, t);
    }

    // ---- Phase C: output projection Out = Z @ Wp^T + bp ----
    gemm_bf16x3<<<gg, 128, 0, stream>>>(Z, Wp, bp, (float*)d_out, 0);
}
